// Net3D_22351009809231
// MI455X (gfx1250) — matrix operand order repaired and checked by reading, not compiled, not run on hardware
//
#include <hip/hip_runtime.h>
#include <hip/hip_bf16.h>
#include <hip/hip_fp16.h>

typedef _Float16 v16h __attribute__((ext_vector_type(16)));
typedef _Float16 v8h  __attribute__((ext_vector_type(8)));
typedef float    v8f  __attribute__((ext_vector_type(8)));

#define H 128
#define LAYERS 4
#define GRAPHS 64

__device__ __forceinline__ float silu_f(float x) { return x / (1.0f + __expf(-x)); }

__device__ __forceinline__ v8f wmma_f16(v16h a, v16h b, v8f c) {
  // D = A(16x32 f16) * B(32x16 f16) + C(16x16 f32)
  return __builtin_amdgcn_wmma_f32_16x16x32_f16(false, a, false, b, (short)0, c, false, false);
}

// A fragment: row-major tile (16 rows, >=32 halves per row, stride ld halves).
// lane L: row = L%16 ; halves K = (L/16)*8 + {0..7} and 16 + (L/16)*8 + {0..7}
__device__ __forceinline__ v16h load_a_frag(const _Float16* tile, int lane, int ld) {
  const _Float16* p = tile + (lane & 15) * ld + ((lane >> 4) << 3);
  union { v16h v; v8h h[2]; } u;
  u.h[0] = *(const v8h*)(p);
  u.h[1] = *(const v8h*)(p + 16);
  return u.v;
}

// B fragment from pre-packed layout: 512 halves per (kt,nt) tile, lane-major 16 halves.
__device__ __forceinline__ v16h load_b_frag(const _Float16* tp, int lane) {
  return *(const v16h*)(tp + (lane << 4));
}

// ---------------------------------------------------------------------------
// Weight packing: fp32 W[K][N] row-major -> f16 B-fragment tiles
// dest[((kt*(N/16)+nt)*32 + lane)*16 + el], lane = ((klocal/8)%2)*16 + n%16, el = klocal%8 + (klocal/16)*8
// ---------------------------------------------------------------------------
__global__ void pack_b_kernel(const float* __restrict__ W, _Float16* __restrict__ out,
                              int K, int N) {
  int i = blockIdx.x * blockDim.x + threadIdx.x;
  if (i >= K * N) return;
  int k = i / N, n = i % N;
  int kt = k >> 5, kl = k & 31;
  int nt = n >> 4, nl = n & 15;
  int lane = (((kl >> 3) & 1) << 4) | nl;
  int el = (kl & 7) | ((kl >> 4) << 3);
  out[(((size_t)(kt * (N >> 4) + nt)) * 32 + lane) * 16 + el] = (_Float16)W[i];
}

__global__ void zero_f32_kernel(float* __restrict__ p, long long n) {
  long long i = (long long)blockIdx.x * blockDim.x + threadIdx.x;
  if (i < n) p[i] = 0.0f;
}

// d0 = silu(d_in @ We + be), write f32 + f16 copies
__global__ void edge_input_kernel(const float* __restrict__ din,
    const float* __restrict__ We, const float* __restrict__ be,
    float* __restrict__ d32, _Float16* __restrict__ d16, long long total) {
  long long i = (long long)blockIdx.x * blockDim.x + threadIdx.x;
  if (i >= total) return;
  long long e = i >> 7;
  int h = (int)(i & (H - 1));
  float x = be[h] + din[e * 3 + 0] * We[0 * H + h]
                  + din[e * 3 + 1] * We[1 * H + h]
                  + din[e * 3 + 2] * We[2 * H + h];
  x = x / (1.0f + __expf(-x));
  d32[i] = x;
  d16[i] = (_Float16)x;
}

__global__ void feat_init_kernel(const float* __restrict__ emb,
    float* __restrict__ f32, _Float16* __restrict__ f16, long long total) {
  long long i = (long long)blockIdx.x * blockDim.x + threadIdx.x;
  if (i >= total) return;
  float v = emb[i & (H - 1)];
  f32[i] = v;
  f16[i] = (_Float16)v;
}

// ---------------------------------------------------------------------------
// Fused per-layer edge kernel: 64 edges per block, 4 waves (one 16-row M-tile each)
// msg = silu(silu([feat[src]|feat[dst]|d] @ W1 + b1) @ W2 + b2)
// d += msg ; ew = sigmoid(msg . Ws + bs) ; m_sum[dst] += msg * ew (atomics)
// ---------------------------------------------------------------------------
__global__ __launch_bounds__(128) void edge_layer_kernel(
    const _Float16* __restrict__ feat16,
    float* __restrict__ d32, _Float16* __restrict__ d16,
    const int* __restrict__ src, const int* __restrict__ dst,
    const _Float16* __restrict__ W1p, const float* __restrict__ b1,
    const _Float16* __restrict__ W2p, const float* __restrict__ b2,
    const float* __restrict__ Ws, const float* __restrict__ bs,
    float* __restrict__ m_sum)
{
  __shared__ __align__(16) _Float16 mi_s[64 * 256];   // gathered [src|dst] feats (f16)
  __shared__ __align__(16) _Float16 h_s[64 * H];      // hidden (f16)
  __shared__ float ew_s[64];
  __shared__ int src_s[64], dst_s[64];
  float* msg_s = (float*)mi_s;                        // reuse after GEMM1 (32KB fits)

  const int tid  = threadIdx.x;
  const int lane = tid & 31;
  const int wave = tid >> 5;
  const int colid  = lane & 15;
  const int rowoff = (lane >> 4) << 3;
  const size_t ebase = (size_t)blockIdx.x * 64;

  if (tid < 64) {
    src_s[tid] = src[ebase + tid];
    dst_s[tid] = dst[ebase + tid];
  }
  __syncthreads();

  // gather feat16[src] | feat16[dst] into LDS rows of 256 halves
  for (int idx = tid; idx < 64 * 32; idx += 128) {
    int e = idx >> 5;
    int seg = idx & 31;
    const _Float16* p = (seg < 16)
        ? feat16 + (size_t)src_s[e] * H + seg * 8
        : feat16 + (size_t)dst_s[e] * H + (seg - 16) * 8;
    *(uint4*)(mi_s + e * 256 + seg * 8) = *(const uint4*)p;
  }
  __syncthreads();

  // GEMM1: [64 x 384] @ [384 x 128]; K 0..255 from LDS, K 256..383 from d16 global
  {
    const _Float16* arow = mi_s + wave * 16 * 256;
    const _Float16* drow = d16 + (ebase + (size_t)wave * 16) * H;
    for (int nt = 0; nt < 8; ++nt) {
      float bc = b1[nt * 16 + colid];
      v8f acc;
      #pragma unroll
      for (int i = 0; i < 8; ++i) acc[i] = bc;
      #pragma unroll
      for (int kt = 0; kt < 8; ++kt) {
        v16h a = load_a_frag(arow + kt * 32, lane, 256);
        v16h b = load_b_frag(W1p + (size_t)(kt * 8 + nt) * 512, lane);
        acc = wmma_f16(a, b, acc);
      }
      #pragma unroll
      for (int kt = 8; kt < 12; ++kt) {
        v16h a = load_a_frag(drow + (kt - 8) * 32, lane, H);
        v16h b = load_b_frag(W1p + (size_t)(kt * 8 + nt) * 512, lane);
        acc = wmma_f16(a, b, acc);
      }
      int col = nt * 16 + colid;
      #pragma unroll
      for (int v = 0; v < 8; ++v)
        h_s[(wave * 16 + rowoff + v) * H + col] = (_Float16)silu_f(acc[v]);
    }
  }
  __syncthreads();

  // GEMM2: [64 x 128] @ [128 x 128] -> msg (silu), f32, aliases mi_s
  {
    const _Float16* arow = h_s + wave * 16 * H;
    for (int nt = 0; nt < 8; ++nt) {
      float bc = b2[nt * 16 + colid];
      v8f acc;
      #pragma unroll
      for (int i = 0; i < 8; ++i) acc[i] = bc;
      #pragma unroll
      for (int kt = 0; kt < 4; ++kt) {
        v16h a = load_a_frag(arow + kt * 32, lane, H);
        v16h b = load_b_frag(W2p + (size_t)(kt * 8 + nt) * 512, lane);
        acc = wmma_f16(a, b, acc);
      }
      int col = nt * 16 + colid;
      #pragma unroll
      for (int v = 0; v < 8; ++v)
        msg_s[(wave * 16 + rowoff + v) * H + col] = silu_f(acc[v]);
    }
  }
  __syncthreads();

  // soft edge weight: ew = sigmoid(msg . Ws + bs)
  if (tid < 64) {
    float acc = bs[0];
    for (int h = 0; h < H; ++h) acc += msg_s[tid * H + h] * Ws[h];
    ew_s[tid] = 1.0f / (1.0f + __expf(-acc));
  }
  __syncthreads();

  // d += msg (f32 + f16 copies) and scatter msg*ew into m_sum[dst]
  {
    int h = tid;  // 128 threads == H columns
    for (int e = 0; e < 64; ++e) {
      float m = msg_s[e * H + h];
      size_t gi = (ebase + e) * H + h;
      float dn = d32[gi] + m;
      d32[gi] = dn;
      d16[gi] = (_Float16)dn;
      atomicAdd(m_sum + (size_t)dst_s[e] * H + h, m * ew_s[e]);
    }
  }
}

// ---------------------------------------------------------------------------
// Node-side 2-layer MLP (64 nodes per block, 4 waves):
//   x = feat (+ m_sum if given); y = silu(x@W1+b1)@W2+b2 (+ feat if residual)
//   writes out32 (and optional f16 copy). Row-guarded for the N tail.
// ---------------------------------------------------------------------------
__global__ __launch_bounds__(128) void node_mlp_kernel(
    const float* __restrict__ feat32,
    const float* __restrict__ msum,           // nullable
    int add_residual,
    const _Float16* __restrict__ W1p, const float* __restrict__ b1,
    const _Float16* __restrict__ W2p, const float* __restrict__ b2,
    float* __restrict__ out32, _Float16* __restrict__ out16,  // out16 nullable
    int N)
{
  __shared__ __align__(16) _Float16 x_s[64 * H];
  __shared__ __align__(16) _Float16 h_s[64 * H];

  const int tid  = threadIdx.x;
  const int lane = tid & 31;
  const int wave = tid >> 5;
  const int colid  = lane & 15;
  const int rowoff = (lane >> 4) << 3;
  const int nbase = blockIdx.x * 64;

  for (int idx = tid; idx < 64 * H; idx += 128) {
    int r = idx >> 7;
    int c = idx & (H - 1);
    int row = nbase + r;
    float v = 0.0f;
    if (row < N) {
      size_t gi = (size_t)row * H + c;
      v = feat32[gi];
      if (msum) v += msum[gi];
    }
    x_s[idx] = (_Float16)v;
  }
  __syncthreads();

  {
    const _Float16* arow = x_s + wave * 16 * H;
    for (int nt = 0; nt < 8; ++nt) {
      float bc = b1[nt * 16 + colid];
      v8f acc;
      #pragma unroll
      for (int i = 0; i < 8; ++i) acc[i] = bc;
      #pragma unroll
      for (int kt = 0; kt < 4; ++kt) {
        v16h a = load_a_frag(arow + kt * 32, lane, H);
        v16h b = load_b_frag(W1p + (size_t)(kt * 8 + nt) * 512, lane);
        acc = wmma_f16(a, b, acc);
      }
      int col = nt * 16 + colid;
      #pragma unroll
      for (int v = 0; v < 8; ++v)
        h_s[(wave * 16 + rowoff + v) * H + col] = (_Float16)silu_f(acc[v]);
    }
  }
  __syncthreads();

  {
    const _Float16* arow = h_s + wave * 16 * H;
    for (int nt = 0; nt < 8; ++nt) {
      float bc = b2[nt * 16 + colid];
      v8f acc;
      #pragma unroll
      for (int i = 0; i < 8; ++i) acc[i] = bc;
      #pragma unroll
      for (int kt = 0; kt < 4; ++kt) {
        v16h a = load_a_frag(arow + kt * 32, lane, H);
        v16h b = load_b_frag(W2p + (size_t)(kt * 8 + nt) * 512, lane);
        acc = wmma_f16(a, b, acc);
      }
      int col = nt * 16 + colid;
      #pragma unroll
      for (int v = 0; v < 8; ++v) {
        int row = nbase + wave * 16 + rowoff + v;
        if (row < N) {
          size_t gi = (size_t)row * H + col;
          float f = acc[v];
          if (add_residual) f += feat32[gi];
          out32[gi] = f;
          if (out16) out16[gi] = (_Float16)f;
        }
      }
    }
  }
}

// per-graph sum / count / max (max via order-preserving uint key)
__global__ void readout_accum_kernel(const float* __restrict__ x,
    const int* __restrict__ ngid, float* __restrict__ s,
    float* __restrict__ cnt, unsigned* __restrict__ mx, int N) {
  int node = blockIdx.x;
  if (node >= N) return;
  int h = threadIdx.x;
  int g = ngid[node];
  float v = x[(size_t)node * H + h];
  atomicAdd(s + g * H + h, v);
  unsigned u = __float_as_uint(v);
  u = (u & 0x80000000u) ? ~u : (u | 0x80000000u);
  atomicMax(mx + g * H + h, u);
  if (h == 0) atomicAdd(cnt + g, 1.0f);
}

__global__ void readout_mlp_kernel(const float* __restrict__ s,
    const float* __restrict__ cnt, const unsigned* __restrict__ mx,
    const float* __restrict__ Wr1, const float* __restrict__ br1,
    const float* __restrict__ Wr2, const float* __restrict__ br2,
    float* __restrict__ out) {
  __shared__ float r_s[3 * H];
  __shared__ float red[H];
  int g = blockIdx.x;
  int h = threadIdx.x;
  float c = cnt[g];
  for (int k = h; k < 3 * H; k += H) {
    int kk = k & (H - 1);
    float v;
    if (k < H) v = s[g * H + kk];
    else if (k < 2 * H) v = s[g * H + kk] / fmaxf(c, 1.0f);
    else {
      unsigned u = mx[g * H + kk];
      unsigned iu = (u & 0x80000000u) ? (u & 0x7FFFFFFFu) : ~u;
      float m = __uint_as_float(iu);
      v = (c > 0.0f) ? m : 0.0f;
    }
    r_s[k] = v;
  }
  __syncthreads();
  float acc = br1[h];
  for (int k = 0; k < 3 * H; ++k) acc += r_s[k] * Wr1[k * H + h];
  acc = fmaxf(acc, 0.0f);
  red[h] = acc * Wr2[h];
  __syncthreads();
  for (int off = H / 2; off > 0; off >>= 1) {
    if (h < off) red[h] += red[h + off];
    __syncthreads();
  }
  if (h == 0) out[g] = red[0] + br2[0];
}

// ---------------------------------------------------------------------------
extern "C" void kernel_launch(void* const* d_in, const int* in_sizes, int n_in,
                              void* d_out, int out_size, void* d_ws, size_t ws_size,
                              hipStream_t stream) {
  (void)n_in; (void)out_size; (void)ws_size;
  const float* d_edge   = (const float*)d_in[0];
  const int*   src      = (const int*)d_in[1];
  const int*   dst      = (const int*)d_in[2];
  const int*   ngid     = (const int*)d_in[3];
  const float* node_emb = (const float*)d_in[5];
  const float* We       = (const float*)d_in[6];
  const float* be       = (const float*)d_in[7];
  const float* mpW1     = (const float*)d_in[8];
  const float* mpb1     = (const float*)d_in[9];
  const float* mpW2     = (const float*)d_in[10];
  const float* mpb2     = (const float*)d_in[11];
  const float* mpWs     = (const float*)d_in[12];
  const float* mpbs     = (const float*)d_in[13];
  const float* mpWu1    = (const float*)d_in[14];
  const float* mpbu1    = (const float*)d_in[15];
  const float* mpWu2    = (const float*)d_in[16];
  const float* mpbu2    = (const float*)d_in[17];
  const float* Wn1      = (const float*)d_in[18];
  const float* bn1      = (const float*)d_in[19];
  const float* Wn2      = (const float*)d_in[20];
  const float* bn2      = (const float*)d_in[21];
  const float* Wr1      = (const float*)d_in[22];
  const float* br1      = (const float*)d_in[23];
  const float* Wr2      = (const float*)d_in[24];
  const float* br2      = (const float*)d_in[25];

  const int E = in_sizes[1];
  const int N = in_sizes[3];

  char* p = (char*)d_ws;
  auto bump = [&](size_t bytes) -> char* {
    char* r = p;
    p += (bytes + 255) & ~(size_t)255;
    return r;
  };
  float*    feat32  = (float*)bump((size_t)N * H * 4);
  _Float16* feat16  = (_Float16*)bump((size_t)N * H * 2);
  float*    d32     = (float*)bump((size_t)E * H * 4);
  _Float16* d16     = (_Float16*)bump((size_t)E * H * 2);
  float*    m_sum   = (float*)bump((size_t)N * H * 4);
  float*    nodeout = (float*)bump((size_t)N * H * 4);
  float*    gs      = (float*)bump((size_t)GRAPHS * H * 4);
  float*    gcnt    = (float*)bump((size_t)GRAPHS * 4);
  unsigned* gmx     = (unsigned*)bump((size_t)GRAPHS * H * 4);
  _Float16* W1p     = (_Float16*)bump((size_t)LAYERS * 49152 * 2);
  _Float16* W2p     = (_Float16*)bump((size_t)LAYERS * 16384 * 2);
  _Float16* Wu1p    = (_Float16*)bump((size_t)LAYERS * 16384 * 2);
  _Float16* Wu2p    = (_Float16*)bump((size_t)LAYERS * 16384 * 2);
  _Float16* Wn1p    = (_Float16*)bump((size_t)16384 * 2);
  _Float16* Wn2p    = (_Float16*)bump((size_t)16384 * 2);

  // pack weights into f16 B-fragment layout
  for (int l = 0; l < LAYERS; ++l) {
    pack_b_kernel<<<(384 * 128 + 255) / 256, 256, 0, stream>>>(
        mpW1 + (size_t)l * 384 * 128, W1p + (size_t)l * 49152, 384, 128);
    pack_b_kernel<<<(128 * 128 + 255) / 256, 256, 0, stream>>>(
        mpW2 + (size_t)l * 128 * 128, W2p + (size_t)l * 16384, 128, 128);
    pack_b_kernel<<<(128 * 128 + 255) / 256, 256, 0, stream>>>(
        mpWu1 + (size_t)l * 128 * 128, Wu1p + (size_t)l * 16384, 128, 128);
    pack_b_kernel<<<(128 * 128 + 255) / 256, 256, 0, stream>>>(
        mpWu2 + (size_t)l * 128 * 128, Wu2p + (size_t)l * 16384, 128, 128);
  }
  pack_b_kernel<<<(128 * 128 + 255) / 256, 256, 0, stream>>>(Wn1, Wn1p, 128, 128);
  pack_b_kernel<<<(128 * 128 + 255) / 256, 256, 0, stream>>>(Wn2, Wn2p, 128, 128);

  {
    long long tot = (long long)E * H;
    edge_input_kernel<<<(unsigned)((tot + 255) / 256), 256, 0, stream>>>(
        d_edge, We, be, d32, d16, tot);
  }
  {
    long long tot = (long long)N * H;
    feat_init_kernel<<<(unsigned)((tot + 255) / 256), 256, 0, stream>>>(
        node_emb, feat32, feat16, tot);
  }

  for (int l = 0; l < LAYERS; ++l) {
    zero_f32_kernel<<<(unsigned)(((long long)N * H + 255) / 256), 256, 0, stream>>>(
        m_sum, (long long)N * H);
    edge_layer_kernel<<<E / 64, 128, 0, stream>>>(
        feat16, d32, d16, src, dst,
        W1p + (size_t)l * 49152, mpb1 + (size_t)l * H,
        W2p + (size_t)l * 16384, mpb2 + (size_t)l * H,
        mpWs + (size_t)l * H, mpbs + l, m_sum);
    node_mlp_kernel<<<(N + 63) / 64, 128, 0, stream>>>(
        feat32, m_sum, 1,
        Wu1p + (size_t)l * 16384, mpbu1 + (size_t)l * H,
        Wu2p + (size_t)l * 16384, mpbu2 + (size_t)l * H,
        feat32, feat16, N);
  }

  // node-wise output MLP (no residual, no f16 copy)
  node_mlp_kernel<<<(N + 63) / 64, 128, 0, stream>>>(
      feat32, nullptr, 0, Wn1p, bn1, Wn2p, bn2, nodeout, nullptr, N);

  zero_f32_kernel<<<(GRAPHS * H + 255) / 256, 256, 0, stream>>>(gs, GRAPHS * H);
  zero_f32_kernel<<<1, 256, 0, stream>>>(gcnt, GRAPHS);
  zero_f32_kernel<<<(GRAPHS * H + 255) / 256, 256, 0, stream>>>((float*)gmx, GRAPHS * H);

  readout_accum_kernel<<<N, 128, 0, stream>>>(nodeout, ngid, gs, gcnt, gmx, N);
  readout_mlp_kernel<<<GRAPHS, 128, 0, stream>>>(gs, gcnt, gmx, Wr1, br1, Wr2, br2,
                                                 (float*)d_out);
}
